// AlignmentModule_46729244180702
// MI455X (gfx1250) — matrix operand order repaired and checked by reading, not compiled, not run on hardware
//
#include <hip/hip_runtime.h>
#include <hip/hip_bf16.h>
#include <math.h>

// ---------------------------------------------------------------------------
// AlignmentModule on gfx1250 (MI455X): all conv1d layers + pairwise distance
// run as wave32 WMMA GEMMs (v_wmma_f32_16x16x32_f16, f32 accum) with 32x64
// register tiles per wave (2x4 fragment blocking -> 8 WMMA per 12 b128 loads).
// GEMM blocks are 128 threads (4 waves = 1 wave/SIMD) with launch_bounds(128,1)
// so the allocator has the full wave32 register file -> no accumulator spills.
// Beta-binomial prior uses an lgamma-of-integers lookup table (all gammaln
// arguments in the reference are integers <= S+F).
// ---------------------------------------------------------------------------

typedef __attribute__((ext_vector_type(16))) _Float16 v16h;
typedef __attribute__((ext_vector_type(8)))  _Float16 h8;
typedef __attribute__((ext_vector_type(8)))  float    v8f;

static constexpr int kB = 8, kS = 128, kF = 512, kD = 256;
static constexpr int kFEAT = 80, kFEATP = 96;           // pad 80 -> 96 (3x32)
static constexpr int kLG = 1024;                        // lgamma table size

// ---- fragment loaders (layouts per CDNA5 ISA 7.12.2, wave32) --------------
__device__ __forceinline__ v16h frag_pair(const _Float16* p0, const _Float16* p1) {
  h8 lo = *(const h8*)p0;
  h8 hi = *(const h8*)p1;
  v16h r;
#pragma unroll
  for (int i = 0; i < 8; ++i) { r[i] = lo[i]; r[i + 8] = hi[i]; }
  return r;
}

// A: 16x32 f16. lane&15 = M row; low/high half-wave covers K octets
// {k0..k0+7, k0+16..k0+23} / {k0+8..k0+15, k0+24..k0+31}.
__device__ __forceinline__ v16h load_afrag(const _Float16* __restrict__ A,
                                           int lda, int m0, int k0, int lane) {
  int m  = m0 + (lane & 15);
  int kb = k0 + ((lane >> 4) << 3);
  const _Float16* p = A + (long)m * lda + kb;
  return frag_pair(p, p + 16);
}

// B: 32x16 f16 stored as B^T (N-major, ldb = K). lane&15 = N col; the 16
// halfs of the fragment are K = kb..kb+7, kb+16..kb+23 with kb = k0 + 8*(lane>=16).
__device__ __forceinline__ v16h load_bfrag(const _Float16* __restrict__ Bt,
                                           int ldb, int n0, int k0, int lane) {
  int n  = n0 + (lane & 15);
  int kb = k0 + ((lane >> 4) << 3);
  const _Float16* p = Bt + (long)n * ldb + kb;
  return frag_pair(p, p + 16);
}

// ---- generic conv-as-GEMM kernel: Out = act(A @ Bt^T + bias) --------------
// One 32x64 register tile per wave (2x4 WMMA sub-tiles); 4 waves per block
// (one wave per SIMD -> full register file); grid.y = batch.
__global__ void __launch_bounds__(128, 1)
k_conv_gemm(const _Float16* __restrict__ Abase, int lda, long astride,
            const _Float16* __restrict__ Bt,
            const float* __restrict__ bias,
            _Float16* __restrict__ Out, int ldo, long ostride,
            int M, int N, int K, int relu) {
  const int lane = threadIdx.x & 31;
  const int wave = threadIdx.x >> 5;
  const int ntiles = N >> 6;                       // 64-wide N tiles
  const int tile = blockIdx.x * 4 + wave;
  if (tile >= (M >> 5) * ntiles) return;
  const int b  = blockIdx.y;
  const int m0 = (tile / ntiles) << 5;
  const int n0 = (tile % ntiles) << 6;

  const _Float16* A = Abase + (long)b * astride;
  _Float16*       O = Out   + (long)b * ostride;

  v8f acc[2][4] = {};
  for (int k0 = 0; k0 < K; k0 += 32) {
    v16h af[2], bf[4];
#pragma unroll
    for (int i = 0; i < 2; ++i) af[i] = load_afrag(A, lda, m0 + 16 * i, k0, lane);
#pragma unroll
    for (int j = 0; j < 4; ++j) bf[j] = load_bfrag(Bt, K, n0 + 16 * j, k0, lane);
#pragma unroll
    for (int i = 0; i < 2; ++i)
#pragma unroll
      for (int j = 0; j < 4; ++j)
        acc[i][j] = __builtin_amdgcn_wmma_f32_16x16x32_f16(
            false, af[i], false, bf[j], (short)0, acc[i][j], false, false);
  }
  // C/D layout: VGPR r, lanes 0-15 -> M=m0+r, lanes 16-31 -> M=m0+r+8
#pragma unroll
  for (int j = 0; j < 4; ++j) {
    const int n = n0 + 16 * j + (lane & 15);
    const float bv = bias[n];
#pragma unroll
    for (int i = 0; i < 2; ++i) {
      const int mb = m0 + 16 * i + ((lane >> 4) << 3);
#pragma unroll
      for (int r = 0; r < 8; ++r) {
        float v = acc[i][j][r] + bv;
        if (relu) v = fmaxf(v, 0.0f);
        O[(long)(mb + r) * ldo + n] = (_Float16)v;
      }
    }
  }
}

// ---- distance GEMM: logit[b,s,f] = mask ? -sqrt(|h|^2+|m|^2-2 h.m) : -inf -
__global__ void __launch_bounds__(128, 1)
k_dist(const _Float16* __restrict__ hh, const _Float16* __restrict__ mm,
       const float* __restrict__ hn, const float* __restrict__ mn,
       const int* __restrict__ tok, float* __restrict__ logit) {
  const int lane = threadIdx.x & 31;
  const int wave = threadIdx.x >> 5;
  const int ntiles = kF >> 6;                      // 8
  const int tile = blockIdx.x * 4 + wave;          // 32 tiles/batch, grid.x=8
  const int b  = blockIdx.y;
  const int m0 = (tile / ntiles) << 5;
  const int n0 = (tile % ntiles) << 6;

  const _Float16* A  = hh + (long)b * kS * kD;
  const _Float16* Bt = mm + (long)b * kF * kD;

  v8f acc[2][4] = {};
  for (int k0 = 0; k0 < kD; k0 += 32) {
    v16h af[2], bf[4];
#pragma unroll
    for (int i = 0; i < 2; ++i) af[i] = load_afrag(A, kD, m0 + 16 * i, k0, lane);
#pragma unroll
    for (int j = 0; j < 4; ++j) bf[j] = load_bfrag(Bt, kD, n0 + 16 * j, k0, lane);
#pragma unroll
    for (int i = 0; i < 2; ++i)
#pragma unroll
      for (int j = 0; j < 4; ++j)
        acc[i][j] = __builtin_amdgcn_wmma_f32_16x16x32_f16(
            false, af[i], false, bf[j], (short)0, acc[i][j], false, false);
  }
  const int L = tok[b];
#pragma unroll
  for (int j = 0; j < 4; ++j) {
    const int f = n0 + 16 * j + (lane & 15);
    const float mnf = mn[b * kF + f];
#pragma unroll
    for (int i = 0; i < 2; ++i) {
      const int mb = m0 + 16 * i + ((lane >> 4) << 3);
#pragma unroll
      for (int r = 0; r < 8; ++r) {
        const int s = mb + r;
        float d2 = hn[b * kS + s] + mnf - 2.0f * acc[i][j][r];
        float lv = (s < L) ? -sqrtf(fmaxf(d2, 0.0f)) : -__builtin_inff();
        logit[((long)b * kS + s) * kF + f] = lv;
      }
    }
  }
}

// ---- prep kernels ---------------------------------------------------------
__global__ void k_lgtab(float* lg) {
  int i = blockIdx.x * blockDim.x + threadIdx.x;
  if (i < kLG) lg[i] = lgammaf((float)i);
}

__global__ void k_prep_h16(const float* __restrict__ h, _Float16* __restrict__ h16) {
  int idx = blockIdx.x * blockDim.x + threadIdx.x;
  const int total = kB * (kS + 2) * kD;
  if (idx >= total) return;
  int d = idx % kD, r = (idx / kD) % (kS + 2), b = idx / (kD * (kS + 2));
  float v = (r >= 1 && r <= kS) ? h[((long)b * kS + (r - 1)) * kD + d] : 0.0f;
  h16[idx] = (_Float16)v;
}

__global__ void k_prep_m16(const float* __restrict__ m, _Float16* __restrict__ m16) {
  int idx = blockIdx.x * blockDim.x + threadIdx.x;
  const int total = kB * (kF + 2) * kFEATP;
  if (idx >= total) return;
  int c = idx % kFEATP, r = (idx / kFEATP) % (kF + 2), b = idx / (kFEATP * (kF + 2));
  float v = (r >= 1 && r <= kF && c < kFEAT)
              ? m[((long)b * kF + (r - 1)) * kFEAT + c] : 0.0f;
  m16[idx] = (_Float16)v;
}

// weights (N, Cin, KW) f32 -> B^T (N, KW*Cinpad) f16, k = tap*Cinpad + c
__global__ void k_prep_w(const float* __restrict__ w, _Float16* __restrict__ bt,
                         int N, int Cin, int KW, int Cinpad) {
  int idx = blockIdx.x * blockDim.x + threadIdx.x;
  const int Kp = KW * Cinpad;
  if (idx >= N * Kp) return;
  int k = idx % Kp, n = idx / Kp;
  int tap = k / Cinpad, c = k % Cinpad;
  float v = (c < Cin) ? w[((long)n * Cin + c) * KW + tap] : 0.0f;
  bt[idx] = (_Float16)v;
}

__global__ void k_zero_pads(_Float16* buf, int rowlen, int M, long bstride) {
  int idx = blockIdx.x * blockDim.x + threadIdx.x;
  const int total = kB * 2 * rowlen;
  if (idx >= total) return;
  int c = idx % rowlen, which = (idx / rowlen) & 1, b = idx / (2 * rowlen);
  long r = which ? (long)(M + 1) : 0;
  buf[(long)b * bstride + r * rowlen + c] = (_Float16)0.0f;
}

__global__ void k_norms(const _Float16* __restrict__ hh, const _Float16* __restrict__ mm,
                        float* __restrict__ hn, float* __restrict__ mn) {
  int idx = blockIdx.x * blockDim.x + threadIdx.x;
  if (idx < kB * kS) {
    const _Float16* p = hh + (long)idx * kD;
    float s = 0.0f;
    for (int d = 0; d < kD; ++d) { float v = (float)p[d]; s += v * v; }
    hn[idx] = s;
  } else if (idx < kB * kS + kB * kF) {
    int j = idx - kB * kS;
    const _Float16* p = mm + (long)j * kD;
    float s = 0.0f;
    for (int d = 0; d < kD; ++d) { float v = (float)p[d]; s += v * v; }
    mn[j] = s;
  }
}

// ---- log-softmax over S + integer-lgamma beta-binomial prior --------------
__global__ void k_post(const float* __restrict__ logit, const float* __restrict__ lg,
                       const int* __restrict__ tok, const int* __restrict__ fl,
                       float* __restrict__ out) {
  int idx = blockIdx.x * blockDim.x + threadIdx.x;   // idx = b*F + f
  if (idx >= kB * kF) return;
  const int b = idx / kF, f = idx % kF;
  const float* col = logit + (long)b * kS * kF + f;

  float mx = -__builtin_inff();
  for (int s = 0; s < kS; ++s) mx = fmaxf(mx, col[(long)s * kF]);
  float sum = 0.0f;
  for (int s = 0; s < kS; ++s) sum += expf(col[(long)s * kF] - mx);
  const float lse = mx + logf(sum);

  const int L = tok[b], Fl = fl[b];
  // constant-per-column part of the prior (valid region only)
  float cc = 0.0f;
  if (f < Fl) cc = lg[L] - lg[L + Fl] + lg[Fl + 1] - lg[f + 1] - lg[Fl - f];
  float* o = out + (long)b * kS * kF + f;
  for (int s = 0; s < kS; ++s) {
    float prior;
    if (s < L && f < Fl) {
      prior = cc - lg[s + 1] - lg[L - s] + lg[s + f + 1] + lg[L + Fl - 1 - s - f];
    } else {
      prior = -__builtin_inff();
    }
    o[(long)s * kF] = prior + (col[(long)s * kF] - lse);
  }
}

// ---------------------------------------------------------------------------
extern "C" void kernel_launch(void* const* d_in, const int* in_sizes, int n_in,
                              void* d_out, int out_size, void* d_ws, size_t ws_size,
                              hipStream_t stream) {
  const float* h   = (const float*)d_in[0];
  const float* m   = (const float*)d_in[1];
  const float* tw1 = (const float*)d_in[2];
  const float* tb1 = (const float*)d_in[3];
  const float* tw2 = (const float*)d_in[4];
  const float* tb2 = (const float*)d_in[5];
  const float* fw1 = (const float*)d_in[6];
  const float* fb1 = (const float*)d_in[7];
  const float* fw2 = (const float*)d_in[8];
  const float* fb2 = (const float*)d_in[9];
  const float* fw3 = (const float*)d_in[10];
  const float* fb3 = (const float*)d_in[11];
  const int*   tok = (const int*)d_in[13];
  const int*   fl  = (const int*)d_in[14];
  float* out = (float*)d_out;

  // ---- carve workspace (~12 MB) ----
  char* p = (char*)d_ws;
  auto carve = [&](size_t bytes) {
    void* r = (void*)p;
    p += (bytes + 255) & ~(size_t)255;
    return r;
  };
  float*    lg    = (float*)   carve(kLG * 4);
  _Float16* h16   = (_Float16*)carve((size_t)kB * (kS + 2) * kD * 2);
  _Float16* m16   = (_Float16*)carve((size_t)kB * (kF + 2) * kFEATP * 2);
  _Float16* x1    = (_Float16*)carve((size_t)kB * (kS + 2) * kD * 2);
  _Float16* y1    = (_Float16*)carve((size_t)kB * (kF + 2) * kD * 2);
  _Float16* y2    = (_Float16*)carve((size_t)kB * (kF + 2) * kD * 2);
  _Float16* hh16  = (_Float16*)carve((size_t)kB * kS * kD * 2);
  _Float16* mm16  = (_Float16*)carve((size_t)kB * kF * kD * 2);
  _Float16* Btw1  = (_Float16*)carve((size_t)kD * (3 * kD) * 2);
  _Float16* Btw2  = (_Float16*)carve((size_t)kD * kD * 2);
  _Float16* Bfw1  = (_Float16*)carve((size_t)kD * (3 * kFEATP) * 2);
  _Float16* Bfw2  = (_Float16*)carve((size_t)kD * (3 * kD) * 2);
  _Float16* Bfw3  = (_Float16*)carve((size_t)kD * kD * 2);
  float*    hn    = (float*)   carve((size_t)kB * kS * 4);
  float*    mn    = (float*)   carve((size_t)kB * kF * 4);
  float*    logit = (float*)   carve((size_t)kB * kS * kF * 4);

  auto cdiv = [](int a, int b) { return (a + b - 1) / b; };

  // ---- prep ----
  k_lgtab<<<cdiv(kLG, 256), 256, 0, stream>>>(lg);
  k_prep_h16<<<cdiv(kB * (kS + 2) * kD, 256), 256, 0, stream>>>(h, h16);
  k_prep_m16<<<cdiv(kB * (kF + 2) * kFEATP, 256), 256, 0, stream>>>(m, m16);
  k_prep_w<<<cdiv(kD * 3 * kD, 256), 256, 0, stream>>>(tw1, Btw1, kD, kD, 3, kD);
  k_prep_w<<<cdiv(kD * kD, 256), 256, 0, stream>>>(tw2, Btw2, kD, kD, 1, kD);
  k_prep_w<<<cdiv(kD * 3 * kFEATP, 256), 256, 0, stream>>>(fw1, Bfw1, kD, kFEAT, 3, kFEATP);
  k_prep_w<<<cdiv(kD * 3 * kD, 256), 256, 0, stream>>>(fw2, Bfw2, kD, kD, 3, kD);
  k_prep_w<<<cdiv(kD * kD, 256), 256, 0, stream>>>(fw3, Bfw3, kD, kD, 1, kD);
  k_zero_pads<<<cdiv(kB * 2 * kD, 256), 256, 0, stream>>>(x1, kD, kS, (long)(kS + 2) * kD);
  k_zero_pads<<<cdiv(kB * 2 * kD, 256), 256, 0, stream>>>(y1, kD, kF, (long)(kF + 2) * kD);
  k_zero_pads<<<cdiv(kB * 2 * kD, 256), 256, 0, stream>>>(y2, kD, kF, (long)(kF + 2) * kD);

  // ---- WMMA GEMM conv stacks (32x64 tiles per wave, 4 waves/block) ----
  // text conv3 (K=768, relu) : h16 -> x1
  {
    dim3 g(cdiv((kS / 32) * (kD / 64), 4), kB);
    k_conv_gemm<<<g, 128, 0, stream>>>(h16, kD, (long)(kS + 2) * kD, Btw1, tb1,
                                       x1 + kD, kD, (long)(kS + 2) * kD,
                                       kS, kD, 3 * kD, 1);
  }
  // text conv1 (K=256)       : x1 -> hh16
  {
    dim3 g(cdiv((kS / 32) * (kD / 64), 4), kB);
    k_conv_gemm<<<g, 128, 0, stream>>>(x1 + kD, kD, (long)(kS + 2) * kD, Btw2, tb2,
                                       hh16, kD, (long)kS * kD,
                                       kS, kD, kD, 0);
  }
  // feat conv3 (K=288, relu) : m16 -> y1
  {
    dim3 g(cdiv((kF / 32) * (kD / 64), 4), kB);
    k_conv_gemm<<<g, 128, 0, stream>>>(m16, kFEATP, (long)(kF + 2) * kFEATP, Bfw1, fb1,
                                       y1 + kD, kD, (long)(kF + 2) * kD,
                                       kF, kD, 3 * kFEATP, 1);
  }
  // feat conv3 (K=768, relu) : y1 -> y2
  {
    dim3 g(cdiv((kF / 32) * (kD / 64), 4), kB);
    k_conv_gemm<<<g, 128, 0, stream>>>(y1, kD, (long)(kF + 2) * kD, Bfw2, fb2,
                                       y2 + kD, kD, (long)(kF + 2) * kD,
                                       kF, kD, 3 * kD, 1);
  }
  // feat conv1 (K=256)       : y2 -> mm16
  {
    dim3 g(cdiv((kF / 32) * (kD / 64), 4), kB);
    k_conv_gemm<<<g, 128, 0, stream>>>(y2 + kD, kD, (long)(kF + 2) * kD, Bfw3, fb3,
                                       mm16, kD, (long)kF * kD,
                                       kF, kD, kD, 0);
  }

  // ---- distance + softmax + prior ----
  k_norms<<<cdiv(kB * kS + kB * kF, 256), 256, 0, stream>>>(hh16, mm16, hn, mn);
  {
    dim3 g(cdiv((kS / 32) * (kF / 64), 4), kB);   // 8 x 8
    k_dist<<<g, 128, 0, stream>>>(hh16, mm16, hn, mn, tok, logit);
  }
  k_post<<<cdiv(kB * kF, 256), 256, 0, stream>>>(logit, lg, tok, fl, out);
}
